// LSTM_32890859552831
// MI455X (gfx1250) — matrix operand order repaired and checked by reading, not compiled, not run on hardware
//
#include <hip/hip_runtime.h>

// ---------------------------------------------------------------------------
// Problem constants (from reference)
// ---------------------------------------------------------------------------
#define VOCAB  32000
#define EMBED  128
#define HIDDEN 256
#define SS     32
#define BS     64
#define MSTL   128
#define MSEL   24
#define QL     16
#define TSTEPS (MSTL + 3)   // story(128) + query(1) + 2 zero pads = 131

typedef __attribute__((ext_vector_type(16))) _Float16 v16h;
typedef __attribute__((ext_vector_type(8)))  _Float16 v8h;
typedef __attribute__((ext_vector_type(8)))  float    v8f;

__device__ __forceinline__ float sigmoidf_(float x) {
    return 1.0f / (1.0f + __expf(-x));
}

// Load one 16x32 f16 A-fragment for wave32 WMMA from a row-major f16 matrix.
// Per ISA (CDNA5 ISA, 16-bit A 16x32): lane l covers row M=l&15,
// halves [kh..kh+7] and [kh+16..kh+23] with kh = kt*32 + (l>>4)*8.
__device__ __forceinline__ v16h load_a_frag(const _Float16* __restrict__ m,
                                            int row_stride_halves,
                                            int row, int kh) {
    const _Float16* p = m + (size_t)row * row_stride_halves + kh;
    v8h lo = *(const v8h*)(p);
    v8h hi = *(const v8h*)(p + 16);
    v16h a;
#pragma unroll
    for (int i = 0; i < 8; ++i) { a[i] = lo[i]; a[i + 8] = hi[i]; }
    return a;
}

// ---------------------------------------------------------------------------
// Pack B = W^T (W is [N][K] f32 row-major) into per-fragment f16 layout.
// Fragment (kt,nt) covers K rows kt*32..+31, N cols nt*16..+15.
// Per ISA B 32x16 layout: lane l holds n = nt*16 + (l&15),
// k = kt*32 + (l>>4)*8 + i (i<8), kt*32 + 16 + (l>>4)*8 + (i-8) (i>=8), stored contiguously so the GEMM
// does ONE aligned 32B v16h load per lane per fragment (1 KB per fragment).
// ---------------------------------------------------------------------------
__global__ void pack_bT_kernel(const float* __restrict__ W,
                               _Float16* __restrict__ out, int N, int K) {
    int idx = blockIdx.x * blockDim.x + threadIdx.x;
    int total = (N * K) >> 1;
    if (idx >= total) return;
    int f = idx >> 8;            // 256 dwords per fragment
    int r = idx & 255;
    int l = r >> 3;              // lane 0..31
    int v = r & 7;               // dword within lane
    int ntiles = N >> 4;
    int kt = f / ntiles;
    int nt = f - kt * ntiles;
    int n = nt * 16 + (l & 15);
    int k = kt * 32 + (l >> 4) * 8 + 2 * v + (v >= 4 ? 8 : 0);
    size_t o = (size_t)f * 512 + (size_t)l * 16 + 2 * v;
    out[o]     = (_Float16)W[(size_t)n * K + k];
    out[o + 1] = (_Float16)W[(size_t)n * K + k + 1];
}

__global__ void bias_combine_kernel(const float* __restrict__ b_ih,
                                    const float* __restrict__ b_hh,
                                    float* __restrict__ bias) {
    int i = blockIdx.x * blockDim.x + threadIdx.x;
    if (i < 4 * HIDDEN) bias[i] = b_ih[i] + b_hh[i];
}

// ---------------------------------------------------------------------------
// Encode: embed + position-encode + sum over words -> x[t][b][e] in f16.
// ---------------------------------------------------------------------------
__global__ void encode_kernel(const int* __restrict__ story,
                              const int* __restrict__ query,
                              const float* __restrict__ table,
                              _Float16* __restrict__ x) {
    int bid = blockIdx.x;                  // = t*BS + b
    int t = bid / BS;
    int b = bid - t * BS;
    int e = threadIdx.x;
    float kfac = (e + 1.0f) * (1.0f / (float)EMBED);
    float acc = 0.0f;
    if (t < MSTL) {
        const int* sp = story + ((size_t)b * MSTL + t) * MSEL;
        for (int w = 0; w < MSEL; ++w) {
            int tok = sp[w];
            float j = (w + 1.0f) * (1.0f / (float)SS);
            float pe = (1.0f - j) - kfac * (1.0f - 2.0f * j);
            acc += table[(size_t)tok * EMBED + e] * pe;
        }
    } else if (t == MSTL) {
        const int* qp = query + (size_t)b * QL;
        for (int w = 0; w < QL; ++w) {
            int tok = qp[w];
            float j = (w + 1.0f) * (1.0f / (float)SS);
            float pe = (1.0f - j) - kfac * (1.0f - 2.0f * j);
            acc += table[(size_t)tok * EMBED + e] * pe;
        }
    }
    x[((size_t)t * BS + b) * EMBED + e] = (_Float16)acc;
}

// ---------------------------------------------------------------------------
// Persistent LSTM: ONE workgroup, 1024 threads = 32 wave32s on one WGP.
// - c state lives in VGPRs across all 131 steps (2 tiles x v8f per wave)
// - h state double-buffered in LDS (f16, rows padded to 264 halves)
// - weights stream from L2 each step (768 KB total, permanently hot).
//   An opaque per-iteration zero (inline asm) is folded into the weight base
//   pointers so LICM cannot hoist fragment loads across the t loop (which
//   previously caused VGPR spills + scratch reloads in the inner loop).
// - one s_barrier per timestep
// Wave w owns pairs p = 2w, 2w+1 (same hidden tile ht = w>>1, batch tiles
// mt = {0,1} or {2,3}); per pair: 4 gate accumulators x 12 K-tiles of
// v_wmma_f32_16x16x32_f16 -> 96 WMMAs/wave/step.
// ---------------------------------------------------------------------------
__global__ __launch_bounds__(1024)
void lstm_kernel(const _Float16* __restrict__ x,     // [T][BS][EMBED]
                 const _Float16* __restrict__ wihp,  // packed frags [4][64]
                 const _Float16* __restrict__ whhp,  // packed frags [8][64]
                 const float* __restrict__ bias,     // [4*HIDDEN]
                 _Float16* __restrict__ h_out) {     // [BS][HIDDEN] f16
    __shared__ alignas(16) _Float16 hbuf[2][BS][HIDDEN + 8];

    const int tid  = threadIdx.x;
    const int lane = tid & 31;
    const int wave = tid >> 5;
    const int nrow = lane & 15;
    const int kgrp = lane >> 4;

    // zero initial h (both buffers; buf 0 is read at t=0)
    for (int i = tid; i < 2 * BS * (HIDDEN + 8); i += 1024)
        (&hbuf[0][0][0])[i] = (_Float16)0.0f;
    __syncthreads();

    float cst[2][8];
#pragma unroll
    for (int p = 0; p < 2; ++p)
#pragma unroll
        for (int v = 0; v < 8; ++v) cst[p][v] = 0.0f;

    int buf = 0;
    for (int t = 0; t < TSTEPS; ++t) {
        // Opaque zero: compiler cannot prove the value, so weight-fragment
        // addresses are loop-variant -> no hoisting/spilling across t.
        int zoff;
        asm volatile("s_mov_b32 %0, 0" : "=s"(zoff));
        const _Float16* wihp_t = wihp + zoff;
        const _Float16* whhp_t = whhp + zoff;
        const _Float16* xt = x + (size_t)t * BS * EMBED;

        // Prefetch next step's x fragment rows (gfx1250 global_prefetch_b8).
        if (t + 1 < TSTEPS)
            __builtin_prefetch(xt + (size_t)BS * EMBED + (size_t)lane * EMBED, 0, 3);

#pragma unroll
        for (int pp = 0; pp < 2; ++pp) {
            const int p  = 2 * wave + pp;
            const int mt = p & 3;
            const int ht = p >> 2;

            v8f acc[4];
#pragma unroll
            for (int g = 0; g < 4; ++g) {
                float bv = bias[g * HIDDEN + ht * 16 + nrow];
#pragma unroll
                for (int v = 0; v < 8; ++v) acc[g][v] = bv;
            }
            // x_t @ W_ih^T : K = 128 (keep rolled: caps live fragments)
#pragma unroll 1
            for (int kt = 0; kt < EMBED / 32; ++kt) {
                v16h a = load_a_frag(xt, EMBED, mt * 16 + nrow, kt * 32 + kgrp * 8);
#pragma unroll
                for (int g = 0; g < 4; ++g) {
                    const _Float16* bp =
                        wihp_t + (size_t)(kt * 64 + (g * 16 + ht)) * 512 + lane * 16;
                    v16h bfrag = *(const v16h*)bp;
                    acc[g] = __builtin_amdgcn_wmma_f32_16x16x32_f16(
                        false, a, false, bfrag, (short)0, acc[g], false, false);
                }
            }
            // h @ W_hh^T : K = 256
#pragma unroll 1
            for (int kt = 0; kt < HIDDEN / 32; ++kt) {
                v16h a = load_a_frag(&hbuf[buf][0][0], HIDDEN + 8,
                                     mt * 16 + nrow, kt * 32 + kgrp * 8);
#pragma unroll
                for (int g = 0; g < 4; ++g) {
                    const _Float16* bp =
                        whhp_t + (size_t)(kt * 64 + (g * 16 + ht)) * 512 + lane * 16;
                    v16h bfrag = *(const v16h*)bp;
                    acc[g] = __builtin_amdgcn_wmma_f32_16x16x32_f16(
                        false, a, false, bfrag, (short)0, acc[g], false, false);
                }
            }
            // elementwise cell update; C/D layout: elem v -> m = v + 8*(lane>>4),
            // n = lane&15 within the tile.
#pragma unroll
            for (int v = 0; v < 8; ++v) {
                float ig = sigmoidf_(acc[0][v]);
                float fg = sigmoidf_(acc[1][v]);
                float gg = tanhf(acc[2][v]);
                float og = sigmoidf_(acc[3][v]);
                float c  = fg * cst[pp][v] + ig * gg;
                cst[pp][v] = c;
                float h = og * tanhf(c);
                hbuf[buf ^ 1][mt * 16 + v + 8 * kgrp][ht * 16 + nrow] = (_Float16)h;
            }
        }
        __syncthreads();
        buf ^= 1;
    }

    // Cooperative copy-out of final h (hbuf[buf] holds h from the last step).
    {
        int row = tid >> 4;            // 0..63
        int ch  = (tid & 15) * 16;     // column chunk of 16 halves (32B)
        v8h lo = *(const v8h*)&hbuf[buf][row][ch];
        v8h hi = *(const v8h*)&hbuf[buf][row][ch + 8];
        *(v8h*)(h_out + (size_t)row * HIDDEN + ch)     = lo;
        *(v8h*)(h_out + (size_t)row * HIDDEN + ch + 8) = hi;
    }
}

// ---------------------------------------------------------------------------
// Output GEMM: out[64,32000] = h_n @ lin_W^T + lin_b, f16 WMMA w/ f32 accum.
// One wave per 16x16 tile: 8000 tiles = 1000 blocks x 8 waves. K=256 -> 8 WMMAs.
// ---------------------------------------------------------------------------
__global__ __launch_bounds__(256)
void outgemm_kernel(const _Float16* __restrict__ h,   // [BS][HIDDEN] f16
                    const _Float16* __restrict__ wp,  // packed lin_W frags
                    const float* __restrict__ linb,   // [VOCAB]
                    float* __restrict__ out) {        // [BS][VOCAB]
    const int wid  = blockIdx.x * 8 + (threadIdx.x >> 5);  // 0..7999
    const int lane = threadIdx.x & 31;
    const int NT   = VOCAB / 16;                           // 2000
    const int nt   = wid % NT;
    const int mt   = wid / NT;                             // 0..3
    const int nrow = lane & 15;
    const int kgrp = lane >> 4;

    v8f acc;
    float bv = linb[nt * 16 + nrow];
#pragma unroll
    for (int v = 0; v < 8; ++v) acc[v] = bv;

#pragma unroll 2
    for (int kt = 0; kt < HIDDEN / 32; ++kt) {
        v16h a = load_a_frag(h, HIDDEN, mt * 16 + nrow, kt * 32 + kgrp * 8);
        const _Float16* bp = wp + (size_t)(kt * NT + nt) * 512 + lane * 16;
        v16h bfrag = *(const v16h*)bp;
        acc = __builtin_amdgcn_wmma_f32_16x16x32_f16(
            false, a, false, bfrag, (short)0, acc, false, false);
    }
#pragma unroll
    for (int v = 0; v < 8; ++v) {
        int m = mt * 16 + v + 8 * kgrp;
        int n = nt * 16 + nrow;
        out[(size_t)m * VOCAB + n] = acc[v];
    }
}

// ---------------------------------------------------------------------------
// Host launcher. Inputs (setup_inputs order):
// 0 story(i32) 1 query(i32) 2 embed_table(f32) 3 W_ih 4 W_hh 5 b_ih 6 b_hh
// 7 lin_W 8 lin_b. Output: [64, 32000] f32.
// ---------------------------------------------------------------------------
extern "C" void kernel_launch(void* const* d_in, const int* in_sizes, int n_in,
                              void* d_out, int out_size, void* d_ws, size_t ws_size,
                              hipStream_t stream) {
    const int*   story = (const int*)d_in[0];
    const int*   query = (const int*)d_in[1];
    const float* table = (const float*)d_in[2];
    const float* W_ih  = (const float*)d_in[3];
    const float* W_hh  = (const float*)d_in[4];
    const float* b_ih  = (const float*)d_in[5];
    const float* b_hh  = (const float*)d_in[6];
    const float* lin_W = (const float*)d_in[7];
    const float* lin_b = (const float*)d_in[8];
    float* out = (float*)d_out;

    char* ws = (char*)d_ws;
    _Float16* xbuf  = (_Float16*)(ws + 0);         // 131*64*128*2  = 2,146,304 B
    _Float16* wihp  = (_Float16*)(ws + 2146304);   // 1024*128*2    =   262,144 B
    _Float16* whhp  = (_Float16*)(ws + 2408448);   // 1024*256*2    =   524,288 B
    _Float16* linp  = (_Float16*)(ws + 2932736);   // 32000*256*2   = 16,384,000 B
    float*    biasc = (float*)   (ws + 19316736);  // 1024*4        =     4,096 B
    _Float16* hnbuf = (_Float16*)(ws + 19320832);  // 64*256*2      =    32,768 B
    (void)in_sizes; (void)n_in; (void)out_size; (void)ws_size;

    // Weight packing (f32 -> swizzled f16 B-fragments) + bias combine
    pack_bT_kernel<<<(4 * HIDDEN * EMBED / 2 + 255) / 256, 256, 0, stream>>>(
        W_ih, wihp, 4 * HIDDEN, EMBED);
    pack_bT_kernel<<<(4 * HIDDEN * HIDDEN / 2 + 255) / 256, 256, 0, stream>>>(
        W_hh, whhp, 4 * HIDDEN, HIDDEN);
    pack_bT_kernel<<<(VOCAB * HIDDEN / 2 + 255) / 256, 256, 0, stream>>>(
        lin_W, linp, VOCAB, HIDDEN);
    bias_combine_kernel<<<4, 256, 0, stream>>>(b_ih, b_hh, biasc);

    // Embedding + position encoding -> x (f16)
    encode_kernel<<<TSTEPS * BS, EMBED, 0, stream>>>(story, query, table, xbuf);

    // Persistent single-WGP LSTM (latency-bound recurrence)
    lstm_kernel<<<1, 1024, 0, stream>>>(xbuf, wihp, whhp, biasc, hnbuf);

    // Output projection (memory-bound on lin_W; f16 halves the traffic)
    outgemm_kernel<<<(4 * (VOCAB / 16)) / 8, 256, 0, stream>>>(
        hnbuf, linp, lin_b, out);
}
